// HybridAttention_48326972014899
// MI455X (gfx1250) — matrix operand order repaired and checked by reading, not compiled, not run on hardware
//
#include <hip/hip_runtime.h>
#include <hip/hip_bf16.h>
#include <math.h>

#define DMODEL 512
#define NHEAD  8
#define HDIM   64
#define SEQ    2048
#define BATCH  4
#define NROWS  (BATCH * SEQ)   // 8192
#define THREED (3 * DMODEL)    // 1536
#define WIN_   256

typedef _Float16 h16;
typedef __attribute__((ext_vector_type(16))) _Float16 v16h;
typedef __attribute__((ext_vector_type(8)))  float    v8f;

// ---------------------------------------------------------------------------
// WMMA helpers (f16 in, f32 accumulate), contiguous k order k = 16*hi + j in A and B
// ---------------------------------------------------------------------------
__device__ __forceinline__ v8f wmma16(v16h a, v16h b, v8f c) {
  return __builtin_amdgcn_wmma_f32_16x16x32_f16(
      /*neg_a=*/false, a, /*neg_b=*/false, b,
      /*c_mod=*/(short)0, c, /*reuse_a=*/false, /*reuse_b=*/false);
}

// A-matrix 16x32 f16: lane<16 holds row M=lane, K = 0..15;
// lane>=16 holds same row's K = 16..31. p = row base + k-chunk base.
__device__ __forceinline__ v16h load_a_frag(const h16* p, int hi) {
  v16h a; const int ko = hi ? 16 : 0;
#pragma unroll
  for (int j = 0; j < 16; ++j) a[j] = p[ko + j];
  return a;
}

// B-matrix 32x16 f16: N = lane%16 across lanes, element j <-> K = j + 16*hi.
// p = column(n) row base + k-chunk base + 16*hi  (16 contiguous halfs).
__device__ __forceinline__ v16h load_b_frag(const h16* p) {
  v16h b;
#pragma unroll
  for (int j = 0; j < 16; ++j) b[j] = p[j];
  return b;
}

// ---------------------------------------------------------------------------
// fp32 -> f16 weight conversion
// ---------------------------------------------------------------------------
__global__ void cvt_f16(const float* __restrict__ in, h16* __restrict__ out, int n) {
  int i = blockIdx.x * blockDim.x + threadIdx.x;
  if (i < n) out[i] = (h16)in[i];
}

// ---------------------------------------------------------------------------
// V transpose: VT[((b*H+h)*HD + d)*SEQ + s] = QKV[(b*SEQ+s)*THREED + 2D + h*HD + d]
// Makes P@V B-fragments contiguous along the key axis.
// ---------------------------------------------------------------------------
__global__ void v_transpose(const h16* __restrict__ QKV, h16* __restrict__ VT) {
  int i = blockIdx.x * blockDim.x + threadIdx.x;   // linear over B*H*HD*SEQ
  int s  = i & (SEQ - 1);
  int t  = i >> 11;                                // (b*H + h)*HD + d
  int d  = t & (HDIM - 1);
  int bh = t >> 6;                                 // b*H + h
  int h  = bh & (NHEAD - 1);
  int b  = bh >> 3;
  VT[i] = QKV[(size_t)(b * SEQ + s) * THREED + 2 * DMODEL + h * HDIM + d];
}

// ---------------------------------------------------------------------------
// LayerNorm (torch-style: unbiased std, eps added to std) -> two f16 outputs
// one block (256 threads) per row of 512
// ---------------------------------------------------------------------------
__global__ void ln2_kernel(const float* __restrict__ x,
                           const float* __restrict__ g1, const float* __restrict__ b1,
                           const float* __restrict__ g2, const float* __restrict__ b2,
                           h16* __restrict__ xl, h16* __restrict__ xg) {
  __shared__ float r1[256], r2[256];
  int row = blockIdx.x, t = threadIdx.x;
  const float* xr = x + (size_t)row * DMODEL;
  float a = xr[t], c = xr[t + 256];
  r1[t] = a + c;
  r2[t] = a * a + c * c;
  __syncthreads();
  for (int o = 128; o > 0; o >>= 1) {
    if (t < o) { r1[t] += r1[t + o]; r2[t] += r2[t + o]; }
    __syncthreads();
  }
  float mean = r1[0] * (1.0f / 512.0f);
  float var  = (r2[0] - 512.0f * mean * mean) * (1.0f / 511.0f);  // ddof=1
  float inv  = 1.0f / (sqrtf(fmaxf(var, 0.0f)) + 1e-6f);          // eps on std
  size_t base = (size_t)row * DMODEL;
  float na = (a - mean) * inv, nc = (c - mean) * inv;
  xl[base + t]       = (h16)(g1[t] * na + b1[t]);
  xl[base + t + 256] = (h16)(g1[t + 256] * nc + b1[t + 256]);
  xg[base + t]       = (h16)(g2[t] * na + b2[t]);
  xg[base + t + 256] = (h16)(g2[t + 256] * nc + b2[t + 256]);
}

// ---------------------------------------------------------------------------
// GEMM: OUT[m,n] = sum_k X[m,k] * W[n,k] + bias[n]  (X:[M,K] W:[N,K], f16 out)
// Register-blocked 16x64 per wave; N,K compile-time so the K-loop fully
// unrolls and the scheduler can software-pipeline loads across WMMAs.
// ---------------------------------------------------------------------------
template <int N, int K>
__global__ void __launch_bounds__(256)
gemm_bias_f16(const h16* __restrict__ X, const h16* __restrict__ W,
              const float* __restrict__ bias, h16* __restrict__ OUT, int M) {
  int gw   = (blockIdx.x * blockDim.x + threadIdx.x) >> 5;
  int lane = threadIdx.x & 31;
  const int ng = N >> 6;                 // 64-wide column groups
  int tm = gw / ng, tg = gw % ng;
  if (tm >= (M >> 4)) return;            // wave-uniform
  int hi = lane >> 4, l16 = lane & 15;

  const h16* xrow = X + (size_t)(tm * 16 + l16) * K;
  const h16* wrow = W + (size_t)(tg * 64 + l16) * K + hi * 16;
  const size_t wstep = (size_t)16 * K;
  v8f acc0 = {}, acc1 = {}, acc2 = {}, acc3 = {};
#pragma unroll
  for (int kk = 0; kk < K; kk += 32) {
    v16h a  = load_a_frag(xrow + kk, hi);
    v16h b0 = load_b_frag(wrow + kk);
    v16h b1 = load_b_frag(wrow + wstep + kk);
    v16h b2 = load_b_frag(wrow + 2 * wstep + kk);
    v16h b3 = load_b_frag(wrow + 3 * wstep + kk);
    acc0 = wmma16(a, b0, acc0);
    acc1 = wmma16(a, b1, acc1);
    acc2 = wmma16(a, b2, acc2);
    acc3 = wmma16(a, b3, acc3);
  }
  int n0 = tg * 64 + l16;
  h16* obase = OUT + (size_t)(tm * 16 + hi * 8) * N + n0;
  float b0 = bias[n0], b1 = bias[n0 + 16], b2 = bias[n0 + 32], b3 = bias[n0 + 48];
#pragma unroll
  for (int r = 0; r < 8; ++r) {
    h16* op = obase + (size_t)r * N;
    op[0]  = (h16)(acc0[r] + b0);
    op[16] = (h16)(acc1[r] + b1);
    op[32] = (h16)(acc2[r] + b2);
    op[48] = (h16)(acc3[r] + b3);
  }
}

// ---------------------------------------------------------------------------
// Flash attention: one 16-query tile per wave for one (b,h).
// __launch_bounds__(128, 1): allow a large VGPR budget so Q/K/V fragments,
// accumulators and softmax state stay resident (no scratch spills); ~4
// waves/SIMD is ample to hide latency in this WMMA+VALU-mixed loop.
// ---------------------------------------------------------------------------
__global__ void __launch_bounds__(128, 1)
flash_attn(const h16* __restrict__ QKV, const h16* __restrict__ VT,
           h16* __restrict__ O, int local_mode) {
  __shared__ h16 plds[4 * 16 * 32];  // per-wave 16x32 P staging
  int gw    = (blockIdx.x * blockDim.x + threadIdx.x) >> 5;
  int wslot = (threadIdx.x >> 5) & 3;
  int lane  = threadIdx.x & 31;
  int hi = lane >> 4, l16 = lane & 15;

  const int QT = SEQ / 16;  // 128
  int qt = gw % QT;
  int hh = (gw / QT) % NHEAD;
  int bb = gw / (QT * NHEAD);
  if (bb >= BATCH) return;  // wave-uniform

  const h16* qp = QKV + (size_t)(bb * SEQ + qt * 16 + l16) * THREED + hh * HDIM;
  v16h qa0 = load_a_frag(qp, hi);        // d = 0..31
  v16h qa1 = load_a_frag(qp + 32, hi);   // d = 32..63

  int ks, kc;
  if (local_mode) {
    int c = qt / (WIN_ / 16);
    ks = (c == 0) ? 0 : (c - 1) * WIN_;
    kc = (c == 0) ? WIN_ : 2 * WIN_;
  } else {
    ks = 0; kc = SEQ;
  }

  float mrow[8], lrow[8];
  v8f o0 = {}, o1 = {}, o2 = {}, o3 = {};
#pragma unroll
  for (int r = 0; r < 8; ++r) { mrow[r] = -INFINITY; lrow[r] = 0.0f; }

  h16* pst = plds + wslot * 512;
  const float scale = 0.125f;  // 1/sqrt(64)
  // VT fragment base: row = (b,h,d = t*16 + l16), col = s (key), contiguous in s
  const h16* vtbase = VT + ((size_t)(bb * NHEAD + hh) * HDIM + l16) * SEQ + hi * 16;

  for (int kk = 0; kk < kc; kk += 32) {
    int skey    = ks + kk;             // key index within sequence
    int keybase = bb * SEQ + skey;     // global row of first key

    // ---- K fragments + scores (2 WMMAs per 16-key tile) ----
    const h16* kp0 = QKV + (size_t)(keybase + l16) * THREED + DMODEL + hh * HDIM + hi * 16;
    const h16* kp1 = QKV + (size_t)(keybase + 16 + l16) * THREED + DMODEL + hh * HDIM + hi * 16;
    v16h kb00 = load_b_frag(kp0);
    v16h kb01 = load_b_frag(kp0 + 32);
    v16h kb10 = load_b_frag(kp1);
    v16h kb11 = load_b_frag(kp1 + 32);
    // V fragments issued early: independent of scores, latency hidden
    // under the softmax VALU block below.
    const h16* vp = vtbase + skey;
    v16h vb0 = load_b_frag(vp);
    v16h vb1 = load_b_frag(vp + (size_t)16 * SEQ);
    v16h vb2 = load_b_frag(vp + (size_t)32 * SEQ);
    v16h vb3 = load_b_frag(vp + (size_t)48 * SEQ);

    v8f s0 = {}, s1 = {};
    s0 = wmma16(qa0, kb00, s0);
    s0 = wmma16(qa1, kb01, s0);
    s1 = wmma16(qa0, kb10, s1);
    s1 = wmma16(qa1, kb11, s1);

    // ---- online softmax update (rows live in 16-lane halves) ----
    float p0[8], p1[8];
#pragma unroll
    for (int r = 0; r < 8; ++r) {
      float a0 = s0[r] * scale, a1 = s1[r] * scale;
      float mx = fmaxf(a0, a1);
      mx = fmaxf(mx, __shfl_xor(mx, 1, 32));
      mx = fmaxf(mx, __shfl_xor(mx, 2, 32));
      mx = fmaxf(mx, __shfl_xor(mx, 4, 32));
      mx = fmaxf(mx, __shfl_xor(mx, 8, 32));
      float mnew  = fmaxf(mrow[r], mx);
      float alpha = __expf(mrow[r] - mnew);
      float e0 = __expf(a0 - mnew);
      float e1 = __expf(a1 - mnew);
      float rs = e0 + e1;
      rs += __shfl_xor(rs, 1, 32);
      rs += __shfl_xor(rs, 2, 32);
      rs += __shfl_xor(rs, 4, 32);
      rs += __shfl_xor(rs, 8, 32);
      lrow[r] = lrow[r] * alpha + rs;
      mrow[r] = mnew;
      o0[r] *= alpha; o1[r] *= alpha; o2[r] *= alpha; o3[r] *= alpha;
      p0[r] = e0; p1[r] = e1;
    }

    // ---- stage P (16x32) through LDS: D-layout -> A-layout ----
#pragma unroll
    for (int r = 0; r < 8; ++r) {
      int m = r + hi * 8;
      pst[m * 32 + l16]      = (h16)p0[r];
      pst[m * 32 + 16 + l16] = (h16)p1[r];
    }
    __asm__ volatile("s_wait_dscnt 0x0" ::: "memory");
    v16h pa = load_a_frag(pst + l16 * 32, hi);

    // ---- P @ V over four 16-wide d tiles ----
    o0 = wmma16(pa, vb0, o0);
    o1 = wmma16(pa, vb1, o1);
    o2 = wmma16(pa, vb2, o2);
    o3 = wmma16(pa, vb3, o3);
  }

  // ---- normalize and store f16 attention output ----
  h16* orow = O + (size_t)(bb * SEQ + qt * 16) * DMODEL + hh * HDIM + l16;
#pragma unroll
  for (int r = 0; r < 8; ++r) {
    int m = r + hi * 8;
    float inv = 1.0f / lrow[r];
    h16* op = orow + (size_t)m * DMODEL;
    op[0]  = (h16)(o0[r] * inv);
    op[16] = (h16)(o1[r] * inv);
    op[32] = (h16)(o2[r] * inv);
    op[48] = (h16)(o3[r] * inv);
  }
}

// ---------------------------------------------------------------------------
// Fused output projection: out = AL@WL^T + bL + AG@WG^T + bG  (fp32 out)
// Register-blocked 16x64 per wave, both GEMMs share the accumulators.
// ---------------------------------------------------------------------------
__global__ void __launch_bounds__(256)
out_proj(const h16* __restrict__ AL, const h16* __restrict__ WL,
         const float* __restrict__ BL,
         const h16* __restrict__ AG, const h16* __restrict__ WG,
         const float* __restrict__ BG,
         float* __restrict__ OUT) {
  int gw   = (blockIdx.x * blockDim.x + threadIdx.x) >> 5;
  int lane = threadIdx.x & 31;
  const int NG = DMODEL / 64;  // 8
  int tm = gw / NG, tg = gw % NG;
  if (tm >= NROWS / 16) return;
  int hi = lane >> 4, l16 = lane & 15;

  const h16* alr = AL + (size_t)(tm * 16 + l16) * DMODEL;
  const h16* agr = AG + (size_t)(tm * 16 + l16) * DMODEL;
  const h16* wlr = WL + (size_t)(tg * 64 + l16) * DMODEL + hi * 16;
  const h16* wgr = WG + (size_t)(tg * 64 + l16) * DMODEL + hi * 16;
  const size_t wstep = (size_t)16 * DMODEL;
  v8f acc0 = {}, acc1 = {}, acc2 = {}, acc3 = {};
#pragma unroll
  for (int kk = 0; kk < DMODEL; kk += 32) {
    v16h al  = load_a_frag(alr + kk, hi);
    v16h bl0 = load_b_frag(wlr + kk);
    v16h bl1 = load_b_frag(wlr + wstep + kk);
    v16h bl2 = load_b_frag(wlr + 2 * wstep + kk);
    v16h bl3 = load_b_frag(wlr + 3 * wstep + kk);
    acc0 = wmma16(al, bl0, acc0);
    acc1 = wmma16(al, bl1, acc1);
    acc2 = wmma16(al, bl2, acc2);
    acc3 = wmma16(al, bl3, acc3);
    v16h ag  = load_a_frag(agr + kk, hi);
    v16h bg0 = load_b_frag(wgr + kk);
    v16h bg1 = load_b_frag(wgr + wstep + kk);
    v16h bg2 = load_b_frag(wgr + 2 * wstep + kk);
    v16h bg3 = load_b_frag(wgr + 3 * wstep + kk);
    acc0 = wmma16(ag, bg0, acc0);
    acc1 = wmma16(ag, bg1, acc1);
    acc2 = wmma16(ag, bg2, acc2);
    acc3 = wmma16(ag, bg3, acc3);
  }
  int n0 = tg * 64 + l16;
  float b0 = BL[n0] + BG[n0];
  float b1 = BL[n0 + 16] + BG[n0 + 16];
  float b2 = BL[n0 + 32] + BG[n0 + 32];
  float b3 = BL[n0 + 48] + BG[n0 + 48];
  float* obase = OUT + (size_t)(tm * 16 + hi * 8) * DMODEL + n0;
#pragma unroll
  for (int r = 0; r < 8; ++r) {
    float* op = obase + (size_t)r * DMODEL;
    op[0]  = acc0[r] + b0;
    op[16] = acc1[r] + b1;
    op[32] = acc2[r] + b2;
    op[48] = acc3[r] + b3;
  }
}

// ---------------------------------------------------------------------------
extern "C" void kernel_launch(void* const* d_in, const int* in_sizes, int n_in,
                              void* d_out, int out_size, void* d_ws, size_t ws_size,
                              hipStream_t stream) {
  (void)in_sizes; (void)n_in; (void)out_size; (void)ws_size;
  const float* x         = (const float*)d_in[0];
  const float* ln_l_g    = (const float*)d_in[1];
  const float* ln_l_b    = (const float*)d_in[2];
  const float* ln_g_g    = (const float*)d_in[3];
  const float* ln_g_b    = (const float*)d_in[4];
  const float* loc_in_w  = (const float*)d_in[5];
  const float* loc_in_b  = (const float*)d_in[6];
  const float* loc_out_w = (const float*)d_in[7];
  const float* loc_out_b = (const float*)d_in[8];
  const float* glb_in_w  = (const float*)d_in[9];
  const float* glb_in_b  = (const float*)d_in[10];
  const float* glb_out_w = (const float*)d_in[11];
  const float* glb_out_b = (const float*)d_in[12];

  char* ws = (char*)d_ws;
  size_t off = 0;
  auto alloc = [&](size_t bytes) {
    void* p = ws + off;
    off += (bytes + 255) & ~(size_t)255;
    return p;
  };
  h16* xl16 = (h16*)alloc((size_t)NROWS * DMODEL * 2);
  h16* xg16 = (h16*)alloc((size_t)NROWS * DMODEL * 2);
  h16* wli  = (h16*)alloc((size_t)THREED * DMODEL * 2);
  h16* wgi  = (h16*)alloc((size_t)THREED * DMODEL * 2);
  h16* wlo  = (h16*)alloc((size_t)DMODEL * DMODEL * 2);
  h16* wgo  = (h16*)alloc((size_t)DMODEL * DMODEL * 2);
  h16* qkvl = (h16*)alloc((size_t)NROWS * THREED * 2);
  h16* qkvg = (h16*)alloc((size_t)NROWS * THREED * 2);
  h16* vtl  = (h16*)alloc((size_t)NROWS * DMODEL * 2);
  h16* vtg  = (h16*)alloc((size_t)NROWS * DMODEL * 2);
  h16* atl  = (h16*)alloc((size_t)NROWS * DMODEL * 2);
  h16* atg  = (h16*)alloc((size_t)NROWS * DMODEL * 2);

  // 1) weights -> f16
  cvt_f16<<<(THREED * DMODEL + 255) / 256, 256, 0, stream>>>(loc_in_w, wli, THREED * DMODEL);
  cvt_f16<<<(THREED * DMODEL + 255) / 256, 256, 0, stream>>>(glb_in_w, wgi, THREED * DMODEL);
  cvt_f16<<<(DMODEL * DMODEL + 255) / 256, 256, 0, stream>>>(loc_out_w, wlo, DMODEL * DMODEL);
  cvt_f16<<<(DMODEL * DMODEL + 255) / 256, 256, 0, stream>>>(glb_out_w, wgo, DMODEL * DMODEL);

  // 2) dual layernorm -> f16 activations
  ln2_kernel<<<NROWS, 256, 0, stream>>>(x, ln_l_g, ln_l_b, ln_g_g, ln_g_b, xl16, xg16);

  // 3) QKV projections (register-blocked WMMA GEMM, 16x64 per wave)
  {
    int waves = (NROWS / 16) * (THREED / 64);  // 12288
    gemm_bias_f16<THREED, DMODEL><<<waves * 32 / 256, 256, 0, stream>>>(
        xl16, wli, loc_in_b, qkvl, NROWS);
    gemm_bias_f16<THREED, DMODEL><<<waves * 32 / 256, 256, 0, stream>>>(
        xg16, wgi, glb_in_b, qkvg, NROWS);
  }

  // 3b) transpose V to key-contiguous layout for PV fragments
  {
    int n = NROWS * DMODEL;  // B*H*HD*SEQ
    v_transpose<<<n / 256, 256, 0, stream>>>(qkvl, vtl);
    v_transpose<<<n / 256, 256, 0, stream>>>(qkvg, vtg);
  }

  // 4) attention: global full + local sliding-window (same flash kernel)
  {
    int waves = BATCH * NHEAD * (SEQ / 16);  // 4096
    flash_attn<<<waves * 32 / 128, 128, 0, stream>>>(qkvg, vtg, atg, 0);
    flash_attn<<<waves * 32 / 128, 128, 0, stream>>>(qkvl, vtl, atl, 1);
  }

  // 5) fused output projections + add -> fp32 out
  {
    int waves = (NROWS / 16) * (DMODEL / 64);  // 4096
    out_proj<<<waves * 32 / 256, 256, 0, stream>>>(atl, wlo, loc_out_b,
                                                   atg, wgo, glb_out_b,
                                                   (float*)d_out);
  }
}
